// ShiftedPillarsConcatentation_BNAct_NP9_52493090292260
// MI455X (gfx1250) — matrix operand order repaired and checked by reading, not compiled, not run on hardware
//
#include <hip/hip_runtime.h>
#include <math.h>

// Problem constants (fixed by reference)
#define B_   16
#define C_   576
#define H_   56
#define W_   56
#define HW_  (H_ * W_)          // 3136
#define NTOT (B_ * HW_)         // 50176

// Tiling
#define MT 64                   // M per block
#define NT 128                  // N per block
#define KT 32                   // K step (one f16 WMMA K)
#define PITCH 40                // LDS row pitch in halves (16B-aligned rows, conflict-breaking)

typedef __attribute__((ext_vector_type(16))) _Float16 v16h;
typedef __attribute__((ext_vector_type(8)))  _Float16 v8h;
typedef __attribute__((ext_vector_type(4)))  _Float16 v4h;
typedef __attribute__((ext_vector_type(8)))  float    v8f;
typedef __attribute__((ext_vector_type(4)))  unsigned int u32x4;
typedef __attribute__((ext_vector_type(8)))  int      i32x8;
typedef __attribute__((ext_vector_type(4)))  int      i32x4;

// TDM availability / arity selection (therock headers ship only with the 6-arg toolchain)
#if __has_include(<hip/amd_detail/amd_gfx1250_TDM.h>)
#define TDM_6ARG 1
#endif
#if defined(__has_builtin)
#if __has_builtin(__builtin_amdgcn_tensor_load_to_lds)
#define USE_TDM 1
#endif
#endif

// ---- Tensor Data Mover: load a 2D f16 tile [tile_d1 rows x tile_d0 cols] from a
// row-major f16 tensor (row stride = stride_d0 elements) into LDS at lds_byte_addr,
// inserting 16B of LDS padding after every 64B row => PITCH-40-half rows.
// D# layout per CDNA5 ISA.
#if defined(USE_TDM)
__device__ __forceinline__ void tdm_load_tile_f16(
    unsigned int lds_byte_addr, const _Float16* gsrc,
    unsigned int tensor_d0, unsigned int tensor_d1,
    unsigned int tile_d0, unsigned int tile_d1, unsigned int stride_d0)
{
  const unsigned long long ga = (unsigned long long)(uintptr_t)gsrc;
  u32x4 g0;
  g0[0] = 1u;                                        // count=1, user descriptor
  g0[1] = lds_byte_addr;                             // lds_addr
  g0[2] = (unsigned int)ga;                          // global_addr[31:0]
  g0[3] = (unsigned int)((ga >> 32) & 0x1FFFFFFu)    // global_addr[56:32]
        | (2u << 30);                                // type = 2 ("image")
  i32x8 g1;
  g1[0] = (int)((1u << 16)                           // data_size = 2 bytes (f16)
        |       (1u << 20)                           // pad_enable
        |       (3u << 22)                           // pad_interval: 16 DWORDs (64B)
        |       (3u << 25));                         // pad_amount: 4 DWORDs (16B)
  g1[1] = (int)(tensor_d0 << 16);                    // tensor_dim0[15:0] @ bits 63:48
  g1[2] = (int)((tensor_d0 >> 16) | (tensor_d1 << 16)); // dim0 hi | dim1 lo
  g1[3] = (int)((tensor_d1 >> 16) | (tile_d0 << 16));   // dim1 hi | tile_dim0
  g1[4] = (int)(tile_d1 & 0xFFFFu);                  // tile_dim1 (tile_dim2 = 0)
  g1[5] = (int)stride_d0;                            // tensor_dim0_stride[31:0]
  g1[6] = 0;                                         // stride hi | dim1_stride lo
  g1[7] = 0;
  i32x4 z4 = {0, 0, 0, 0};
#if defined(TDM_6ARG)
  i32x8 z8 = {0, 0, 0, 0, 0, 0, 0, 0};
  __builtin_amdgcn_tensor_load_to_lds(g0, g1, z4, z4, z8, 0);
#else
  __builtin_amdgcn_tensor_load_to_lds(g0, g1, z4, z4, 0);
#endif
}
#endif

// ---- WMMA fragment loaders (CDNA5 wave32 layouts, CDNA5 ISA §7.12.2) ----
__device__ __forceinline__ v16h frag_a(const _Float16* As, int row, int lane) {
  const int kb = (lane & 16) ? 8 : 0;
  v8h lo = *(const v8h*)(As + row * PITCH + kb);
  v8h hi = *(const v8h*)(As + row * PITCH + kb + 16);
  v16h a;
#pragma unroll
  for (int i = 0; i < 8; ++i) { a[i] = lo[i]; a[i + 8] = hi[i]; }
  return a;
}

__device__ __forceinline__ v16h frag_b(const _Float16* Bs, int col, int lane) {
  const int kb = (lane & 16) ? 8 : 0;
  v8h lo = *(const v8h*)(Bs + col * PITCH + kb);
  v8h hi = *(const v8h*)(Bs + col * PITCH + kb + 16);
  v16h b;
#pragma unroll
  for (int i = 0; i < 8; ++i) { b[i] = lo[i]; b[i + 8] = hi[i]; }
  return b;
}

// Shift offsets for block j (derived from _pshift: dh = hs-pt, dw = ws-pl)
__device__ __forceinline__ void shift_off(int j, int s, int& dh, int& dw) {
  switch (j) {
    default:
    case 0: dh = 0;     dw = 0;  break;  // xc
    case 1: dh = 0;     dw = 0;  break;  // xt
    case 2: dh = 1 - s; dw = 0;  break;  // xb
    case 3: dh = 0;     dw = -s; break;  // xr
    case 4: dh = 0;     dw = 1;  break;  // xl
    case 5: dh = 1;     dw = -s; break;  // xtr
    case 6: dh = 1;     dw = 1;  break;  // xtl
    case 7: dh = -s;    dw = -s; break;  // xbr
    case 8: dh = -s;    dw = 1;  break;  // xbl
  }
}

// ======================= Kernel 0: one-time f32 -> f16 weight convert =======================
__global__ __launch_bounds__(256) void k0_convert_weights(
    const float* __restrict__ W9, const float* __restrict__ Wf,
    _Float16* __restrict__ W9h, _Float16* __restrict__ Wfh)
{
  const int i = blockIdx.x * 256 + threadIdx.x;   // 0 .. 576*576-1
  W9h[i] = (_Float16)W9[i];
  Wfh[i] = (_Float16)Wf[i];
}

// ============================ Kernel 1: BN + GELU + GEMM1 ============================
__global__ __launch_bounds__(256) void k1_bn_gelu_gemm(
    const float* __restrict__ x, const float* __restrict__ gamma,
    const float* __restrict__ beta, const float* __restrict__ mean,
    const float* __restrict__ var, const _Float16* __restrict__ W9h,
    const float* __restrict__ b9, _Float16* __restrict__ yws)
{
  __shared__ alignas(16) _Float16 As[MT * PITCH];
  __shared__ alignas(16) _Float16 Bs[NT * PITCH];

  const int t    = threadIdx.x;
  const int lane = t & 31;
  const int wv   = t >> 5;
  const int wm   = wv >> 1;
  const int wn   = wv & 1;
  const int m0   = blockIdx.y * MT;
  const int n0   = blockIdx.x * NT;

  const int brg  = t >> 5;          // B loader: row group 0..7
  const int bnq  = t & 31;          // chunk-of-4 along N

  v8f acc[4] = {};

  for (int c0 = 0; c0 < C_; c0 += KT) {
    // --- A tile via Tensor Data Mover: 64x32 f16 tile of W9h, LDS-padded to PITCH rows ---
#if defined(USE_TDM)
    if (wv == 0) {
      tdm_load_tile_f16((unsigned int)(uintptr_t)As, W9h + m0 * C_ + c0,
                        C_, C_, KT, MT, C_);
    }
#else
    {
      const int arow = t >> 3, aq = t & 7;
#pragma unroll
      for (int p = 0; p < 2; ++p) {
        const int row = p * 32 + arow;
        v4h h = *(const v4h*)(W9h + (m0 + row) * C_ + c0 + aq * 4);
        *(v4h*)(As + row * PITCH + aq * 4) = h;
      }
    }
#endif
    // --- B tile: BN + exact GELU of x, rows c0..c0+31, cols n0..n0+127 (stored [n][k]) ---
#pragma unroll
    for (int p = 0; p < 4; ++p) {
      const int r = p * 8 + brg;
      const int c = c0 + r;
      const float sc = gamma[c] * rsqrtf(var[c] + 1e-5f);
      const float sh = beta[c] - mean[c] * sc;
      const int n  = n0 + bnq * 4;
      const int b  = n / HW_;
      const int hw = n - b * HW_;
      const float* xp = x + (b * C_ + c) * HW_ + hw;
      const float4 f = *(const float4*)xp;
      if (c0 + KT < C_) __builtin_prefetch(xp + KT * HW_, 0, 3);  // near-scope prefetch
      float v[4] = { f.x, f.y, f.z, f.w };
#pragma unroll
      for (int i = 0; i < 4; ++i) {
        const float u = v[i] * sc + sh;
        const float g = 0.5f * u * (1.0f + erff(u * 0.70710678118654752f));
        Bs[(bnq * 4 + i) * PITCH + r] = (_Float16)g;
      }
    }
#if defined(USE_TDM)
    if (wv == 0) __builtin_amdgcn_s_wait_tensorcnt(0);
#endif
    __syncthreads();

    const v16h a = frag_a(As, wm * 16 + (lane & 15), lane);
#pragma unroll
    for (int su = 0; su < 4; ++su) {
      const v16h bb = frag_b(Bs, wn * 64 + su * 16 + (lane & 15), lane);
      acc[su] = __builtin_amdgcn_wmma_f32_16x16x32_f16(
          false, a, false, bb, (short)0, acc[su], false, false);
    }
    __syncthreads();
  }

  // --- epilogue: + b9, f32->f16, scatter to yws[((j*16+b)*64+co)*HW + hw] ---
  const int mrow0 = m0 + wm * 16 + ((lane & 16) >> 1);
  const int ncol0 = n0 + wn * 64 + (lane & 15);
  float bias[8];
#pragma unroll
  for (int r = 0; r < 8; ++r) bias[r] = b9[mrow0 + r];
#pragma unroll
  for (int su = 0; su < 4; ++su) {
    const int n  = ncol0 + su * 16;
    const int b  = n / HW_;
    const int hw = n - b * HW_;
#pragma unroll
    for (int r = 0; r < 8; ++r) {
      const int m  = mrow0 + r;
      const int j  = m >> 6, co = m & 63;
      yws[((j * B_ + b) * 64 + co) * HW_ + hw] = (_Float16)(acc[su][r] + bias[r]);
    }
  }
}

// ============================ Kernel 2: shifted-gather + GEMM2 ============================
__global__ __launch_bounds__(256) void k2_shift_gemm(
    const _Float16* __restrict__ yws, const _Float16* __restrict__ Wfh,
    const float* __restrict__ bf, const int* __restrict__ stepp,
    float* __restrict__ out)
{
  __shared__ alignas(16) _Float16 As[MT * PITCH];
  __shared__ alignas(16) _Float16 Bs[NT * PITCH];

  const int t    = threadIdx.x;
  const int lane = t & 31;
  const int wv   = t >> 5;
  const int wm   = wv >> 1;
  const int wn   = wv & 1;
  const int m0   = blockIdx.y * MT;
  const int n0   = blockIdx.x * NT;

  const int brg  = t >> 5;
  const int bnq  = t & 31;

  const int s = stepp[0];

  v8f acc[4] = {};

  for (int c0 = 0; c0 < C_; c0 += KT) {
    // --- A tile via TDM: 64x32 f16 tile of Wfh ---
#if defined(USE_TDM)
    if (wv == 0) {
      tdm_load_tile_f16((unsigned int)(uintptr_t)As, Wfh + m0 * C_ + c0,
                        C_, C_, KT, MT, C_);
    }
#else
    {
      const int arow = t >> 3, aq = t & 7;
#pragma unroll
      for (int p = 0; p < 2; ++p) {
        const int row = p * 32 + arow;
        v4h h = *(const v4h*)(Wfh + (m0 + row) * C_ + c0 + aq * 4);
        *(v4h*)(As + row * PITCH + aq * 4) = h;
      }
    }
#endif
    // --- B tile: shifted gather of f16 y; shift uniform over this K step ---
    const int j = c0 >> 6;
    int dh, dw; shift_off(j, s, dh, dw);
#pragma unroll
    for (int p = 0; p < 4; ++p) {
      const int r  = p * 8 + brg;
      const int co = (c0 & 63) + r;
      const int n  = n0 + bnq * 4;
      const int b  = n / HW_;
      int hw = n - b * HW_;
      int h = hw / W_;
      int w = hw - h * W_;
      const _Float16* yb = yws + ((j * B_ + b) * 64 + co) * HW_;
#pragma unroll
      for (int i = 0; i < 4; ++i) {
        const int h2 = h + dh, w2 = w + dw;
        _Float16 val = (_Float16)0.f;
        if ((unsigned)h2 < (unsigned)H_ && (unsigned)w2 < (unsigned)W_)
          val = yb[h2 * W_ + w2];
        Bs[(bnq * 4 + i) * PITCH + r] = val;
        if (++w == W_) { w = 0; ++h; }
      }
    }
#if defined(USE_TDM)
    if (wv == 0) __builtin_amdgcn_s_wait_tensorcnt(0);
#endif
    __syncthreads();

    const v16h a = frag_a(As, wm * 16 + (lane & 15), lane);
#pragma unroll
    for (int su = 0; su < 4; ++su) {
      const v16h bb = frag_b(Bs, wn * 64 + su * 16 + (lane & 15), lane);
      acc[su] = __builtin_amdgcn_wmma_f32_16x16x32_f16(
          false, a, false, bb, (short)0, acc[su], false, false);
    }
    __syncthreads();
  }

  // --- epilogue: + bf, f32 store ---
  const int mrow0 = m0 + wm * 16 + ((lane & 16) >> 1);
  const int ncol0 = n0 + wn * 64 + (lane & 15);
  float bias[8];
#pragma unroll
  for (int r = 0; r < 8; ++r) bias[r] = bf[mrow0 + r];
#pragma unroll
  for (int su = 0; su < 4; ++su) {
    const int n  = ncol0 + su * 16;
    const int b  = n / HW_;
    const int hw = n - b * HW_;
#pragma unroll
    for (int r = 0; r < 8; ++r) {
      const int m = mrow0 + r;
      out[(b * C_ + m) * HW_ + hw] = acc[su][r] + bias[r];
    }
  }
}

extern "C" void kernel_launch(void* const* d_in, const int* in_sizes, int n_in,
                              void* d_out, int out_size, void* d_ws, size_t ws_size,
                              hipStream_t stream) {
  const float* x     = (const float*)d_in[0];
  const float* gamma = (const float*)d_in[1];
  const float* beta  = (const float*)d_in[2];
  const float* mean  = (const float*)d_in[3];
  const float* var   = (const float*)d_in[4];
  const float* W9    = (const float*)d_in[5];   // [9,64,576] -> [576][576] (m = k*64+o)
  const float* b9    = (const float*)d_in[6];   // [576]
  const float* Wf    = (const float*)d_in[7];   // [576,576]
  const float* bf    = (const float*)d_in[8];   // [576]
  const int*   step  = (const int*)d_in[9];
  float* out = (float*)d_out;

  // Workspace layout: yws f16 (9*16*64*3136), then W9h, then Wfh
  char* ws = (char*)d_ws;
  _Float16* yws = (_Float16*)ws;
  const size_t ybytes = (size_t)9 * B_ * 64 * HW_ * sizeof(_Float16);
  _Float16* W9h = (_Float16*)(ws + ybytes);
  _Float16* Wfh = W9h + (size_t)C_ * C_;

  k0_convert_weights<<<dim3((C_ * C_) / 256), 256, 0, stream>>>(W9, Wf, W9h, Wfh);

  dim3 grid(NTOT / NT, C_ / MT);                // (392, 9)
  k1_bn_gelu_gemm<<<grid, 256, 0, stream>>>(x, gamma, beta, mean, var, W9h, b9, yws);
  k2_shift_gemm<<<grid, 256, 0, stream>>>(yws, Wfh, bf, step, out);
}
